// DeformConvAtt_11974368821923
// MI455X (gfx1250) — matrix operand order repaired and checked by reading, not compiled, not run on hardware
//
#include <hip/hip_runtime.h>
#include <hip/hip_bf16.h>
#include <cmath>

typedef __attribute__((ext_vector_type(16))) _Float16 v16h;
typedef __attribute__((ext_vector_type(8)))  float    v8f;
typedef __attribute__((ext_vector_type(8)))  _Float16 h8;

static __device__ __forceinline__ v8f wmma_f16(v16h a, v16h b, v8f c) {
  // D = A(16x32,f16) * B(32x16,f16) + C(16x16,f32)
  return __builtin_amdgcn_wmma_f32_16x16x32_f16(false, a, false, b, (short)0, c,
                                                false, false);
}

// ---- Fragment loaders (wave32; contiguous k order k = 16*(lane/16) + s, same in A and B) ----------------
// A 16x32 f16: lanes 0-15 -> M=lane, K=0..15; lanes 16-31 -> M=lane-16, K=16..31
// elemA(m,k) = base[m*ld + k]   (row-major A; one contiguous 32B load/lane)
static __device__ __forceinline__ v16h ldA_rows(const _Float16* __restrict__ base,
                                                int ld, int m0, int k0, int lane) {
  const _Float16* p = base + (size_t)(m0 + (lane & 15)) * ld + (k0 + ((lane >> 4) << 4));
  return *(const v16h*)p;
}

// B 32x16 f16: lanes 0-15 -> N=lane, K=0..15; lanes 16-31 -> N=lane-16, K=16..31
// elemB(k,n) = base[n*ld + k]   (one contiguous 32B load per lane)
static __device__ __forceinline__ v16h ldB_rowsT(const _Float16* __restrict__ base,
                                                 int ld, int n0, int k0, int lane) {
  const _Float16* p = base + (size_t)(n0 + (lane & 15)) * ld + (k0 + ((lane >> 4) << 4));
  return *(const v16h*)p;
}

// ---- Elementwise prep -------------------------------------------------------
__global__ void cvt_f32_f16(const float* __restrict__ src, _Float16* __restrict__ dst,
                            size_t n) {
  size_t i = (size_t)blockIdx.x * blockDim.x + threadIdx.x;
  size_t st = (size_t)gridDim.x * blockDim.x;
  for (; i < n; i += st) dst[i] = (_Float16)src[i];
}

// dst[img][w][h] = src[img][h][w]   (also used for per-batch [C][N] -> [N][C])
__global__ void transpose2d(const _Float16* __restrict__ src, _Float16* __restrict__ dst,
                            int nimg, int H, int W) {
  size_t n = (size_t)nimg * H * W;
  size_t i = (size_t)blockIdx.x * blockDim.x + threadIdx.x;
  size_t st = (size_t)gridDim.x * blockDim.x;
  for (; i < n; i += st) {
    int hw = (int)(i % (size_t)(H * W));
    size_t img = i / (size_t)(H * W);
    int h = hw / W, w = hw - h * W;
    dst[img * (size_t)(H * W) + (size_t)w * H + h] = src[i];
  }
}

// ---- Spatial gram via the transposed view: all-contiguous loads -------------
// G[m,n] = sum_{c} sum_{j<S} Y[(c*S+m)*S + j] * Y[(c*S+n)*S + j],  Y=[C*S,S] row-major.
// (For the W-axis gram pass Y = x_transposed; for the H-axis gram pass Y = x.)
__global__ void gram_spatial(const _Float16* __restrict__ Y, float* __restrict__ G,
                             int C, int S) {
  __shared__ float red[8][32][8];
  int lane = threadIdx.x & 31, wave = threadIdx.x >> 5;
  int m0 = blockIdx.x * 16, n0 = blockIdx.y * 16, b = blockIdx.z;
  const _Float16* Yb = Y + (size_t)b * C * S * S;
  int jsteps = S >> 5;                 // 32-wide K chunks per row (1 or 2)
  int steps = C * jsteps;
  v8f acc = {0.f, 0.f, 0.f, 0.f, 0.f, 0.f, 0.f, 0.f};
  for (int idx = wave; idx < steps; idx += 8) {
    int c = idx / jsteps;
    int jc = (idx - c * jsteps) << 5;
    const _Float16* base = Yb + (size_t)c * S * S + jc;
    v16h a  = ldA_rows (base, S, m0, 0, lane);
    v16h bf = ldB_rowsT(base, S, n0, 0, lane);
    acc = wmma_f16(a, bf, acc);
  }
#pragma unroll
  for (int r = 0; r < 8; ++r) red[wave][lane][r] = acc[r];
  __syncthreads();
  if (wave == 0) {
    float* Gb = G + (size_t)b * S * S;
    int n = n0 + (lane & 15);
    int mb = m0 + ((lane >> 4) << 3);
#pragma unroll
    for (int r = 0; r < 8; ++r) {
      float v = 0.f;
#pragma unroll
      for (int w2 = 0; w2 < 8; ++w2) v += red[w2][lane][r];
      Gb[(size_t)(mb + r) * S + n] = v;
    }
  }
}

// ---- G[c,d] = sum_n X[c,n]*X[d,n],  X=[C,N] row-major (both operands contiguous)
__global__ void gram_xxt(const _Float16* __restrict__ X, float* __restrict__ G,
                         int C, int N) {
  int lane = threadIdx.x & 31;
  int m0 = blockIdx.x * 16, n0 = blockIdx.y * 16, b = blockIdx.z;
  const _Float16* Xb = X + (size_t)b * C * N;
  v8f acc = {0.f, 0.f, 0.f, 0.f, 0.f, 0.f, 0.f, 0.f};
  for (int k0 = 0; k0 < N; k0 += 32) {
    v16h a  = ldA_rows (Xb, N, m0, k0, lane);
    v16h bf = ldB_rowsT(Xb, N, n0, k0, lane);
    acc = wmma_f16(a, bf, acc);
  }
  float* Gb = G + (size_t)b * C * C;
  int n = n0 + (lane & 15);
  int mb = m0 + ((lane >> 4) << 3);
#pragma unroll
  for (int r = 0; r < 8; ++r) Gb[(size_t)(mb + r) * C + n] = acc[r];
}

// ---- row softmax of sgn*G, fp32 math, f16 output ----------------------------
// sgn=+1 spatial, -1 channel (softmax(rowmax-G) == softmax(-G), stabilized).
// transposed=1 writes attn^T (so the spatial apply's B-operand loads contiguously).
__global__ void softmax_rows(const float* __restrict__ G, _Float16* __restrict__ attn,
                             int M, float sgn, int transposed) {
  __shared__ float sred[256];
  int row = blockIdx.x, tid = threadIdx.x;
  int b = row / M, rloc = row - b * M;
  const float* g = G + (size_t)row * M;
  _Float16* ob = attn + (size_t)b * M * M;
  float mx = -3.4e38f;
  for (int j = tid; j < M; j += 256) mx = fmaxf(mx, sgn * g[j]);
  sred[tid] = mx; __syncthreads();
  for (int st = 128; st > 0; st >>= 1) {
    if (tid < st) sred[tid] = fmaxf(sred[tid], sred[tid + st]);
    __syncthreads();
  }
  mx = sred[0]; __syncthreads();
  float sum = 0.f;
  for (int j = tid; j < M; j += 256) sum += __expf(sgn * g[j] - mx);
  sred[tid] = sum; __syncthreads();
  for (int st = 128; st > 0; st >>= 1) {
    if (tid < st) sred[tid] += sred[tid + st];
    __syncthreads();
  }
  float inv = 1.0f / sred[0];
  for (int j = tid; j < M; j += 256) {
    _Float16 v = (_Float16)(__expf(sgn * g[j] - mx) * inv);
    if (transposed) ob[(size_t)j * M + rloc] = v;
    else            ob[(size_t)rloc * M + j] = v;
  }
}

// ---- Y[m,n] = sum_w X[m,w]*attn[w,n]; attnT passed pre-transposed -----------
// mode0: out = 4*base + s*Y (same layout);  mode1 (h<->w input): transposed store.
__global__ void apply_spatial(const _Float16* __restrict__ X,
                              const _Float16* __restrict__ attnT,
                              const float* __restrict__ base, float* __restrict__ out,
                              const float* __restrict__ scale_p, int K, int S, int mode) {
  int lane = threadIdx.x & 31;
  int m0 = blockIdx.x * 16, n0 = blockIdx.y * 16, b = blockIdx.z;
  const _Float16* Xb = X + (size_t)b * K * S;
  const _Float16* Ab = attnT + (size_t)b * S * S;
  v8f acc = {0.f, 0.f, 0.f, 0.f, 0.f, 0.f, 0.f, 0.f};
  for (int k0 = 0; k0 < S; k0 += 32) {
    v16h a  = ldA_rows (Xb, S, m0, k0, lane);
    v16h bf = ldB_rowsT(Ab, S, n0, k0, lane);   // attnT[n][k] == attn[k][n]
    acc = wmma_f16(a, bf, acc);
  }
  float scale = *scale_p;
  size_t obase = (size_t)b * K * S;
  int n = n0 + (lane & 15);
  int mb = m0 + ((lane >> 4) << 3);
#pragma unroll
  for (int r = 0; r < 8; ++r) {
    int m = mb + r;
    if (mode == 0) {
      size_t idx = obase + (size_t)m * S + n;
      out[idx] = 4.0f * base[idx] + scale * acc[r];
    } else {
      int cc = m / S;            // square images: rows-per-image == S
      int rr = m - cc * S;
      size_t idx = obase + (size_t)cc * S * S + (size_t)n * S + rr;
      out[idx] += scale * acc[r];
    }
  }
}

// ---- out[c,n] += sA*(attnA@X)[c,n] + sB*(attnB@X)[c,n] ----------------------
// XT = per-batch [N,C] transposed features -> contiguous B-fragment loads.
__global__ void apply_channel(const _Float16* __restrict__ attnA,
                              const _Float16* __restrict__ attnB,
                              const _Float16* __restrict__ XT, float* __restrict__ out,
                              const float* __restrict__ sA_p, const float* __restrict__ sB_p,
                              int C, int N) {
  int lane = threadIdx.x & 31;
  int m0 = blockIdx.x * 16, n0 = blockIdx.y * 16, b = blockIdx.z;
  const _Float16* Xb = XT + (size_t)b * C * N;
  const _Float16* Aa = attnA + (size_t)b * C * C;
  const _Float16* Ab = attnB + (size_t)b * C * C;
  v8f acc1 = {0.f, 0.f, 0.f, 0.f, 0.f, 0.f, 0.f, 0.f};
  v8f acc2 = {0.f, 0.f, 0.f, 0.f, 0.f, 0.f, 0.f, 0.f};
  for (int k0 = 0; k0 < C; k0 += 32) {
    v16h bf = ldB_rowsT(Xb, C, n0, k0, lane);   // XT[n][d] == X[d][n]
    v16h a1 = ldA_rows (Aa, C, m0, k0, lane);
    acc1 = wmma_f16(a1, bf, acc1);
    v16h a2 = ldA_rows (Ab, C, m0, k0, lane);
    acc2 = wmma_f16(a2, bf, acc2);
  }
  float sA = *sA_p, sB = *sB_p;
  size_t obase = (size_t)b * C * N;
  int n = n0 + (lane & 15);
  int mb = m0 + ((lane >> 4) << 3);
#pragma unroll
  for (int r = 0; r < 8; ++r)
    out[obase + (size_t)(mb + r) * N + n] += sA * acc1[r] + sB * acc2[r];
}

// ---------------------------------------------------------------------------
extern "C" void kernel_launch(void* const* d_in, const int* in_sizes, int n_in,
                              void* d_out, int out_size, void* d_ws, size_t ws_size,
                              hipStream_t stream) {
  (void)in_sizes; (void)n_in; (void)out_size; (void)ws_size;
  const int B = 8, C = 512, TW = 32, SW = 64;
  const int tNpix = TW * TW, sNpix = SW * SW;
  const size_t tN = (size_t)B * C * tNpix;   // 4,194,304
  const size_t sN = (size_t)B * C * sNpix;   // 16,777,216

  const float* t    = (const float*)d_in[0];
  const float* s    = (const float*)d_in[1];
  const float* sc_a = (const float*)d_in[2];
  const float* sc_b = (const float*)d_in[3];
  const float* sc_c = (const float*)d_in[4];
  const float* sc_d = (const float*)d_in[5];
  const float* sc_e = (const float*)d_in[6];
  const float* sc_f = (const float*)d_in[7];
  const float* sc_g = (const float*)d_in[8];
  const float* sc_h = (const float*)d_in[9];

  float* out_t = (float*)d_out;
  float* out_s = out_t + tN;

  char* wp = (char*)d_ws;
  auto bump = [&](size_t bytes) -> char* {
    char* r = wp; wp += (bytes + 255) & ~(size_t)255; return r;
  };
  _Float16* t16  = (_Float16*)bump(tN * 2);                 // [B][C][H][W]
  _Float16* s16  = (_Float16*)bump(sN * 2);
  _Float16* tT16 = (_Float16*)bump(tN * 2);                 // [B][C][W][H]
  _Float16* sT16 = (_Float16*)bump(sN * 2);
  _Float16* tCt  = (_Float16*)bump(tN * 2);                 // [B][N][C]
  _Float16* sCt  = (_Float16*)bump(sN * 2);
  _Float16* aCt  = (_Float16*)bump((size_t)B * C * C * 2);  // channel attn (row-major)
  _Float16* aCs  = (_Float16*)bump((size_t)B * C * C * 2);
  _Float16* aWt  = (_Float16*)bump((size_t)B * tNpix * 2);  // spatial attn (transposed)
  _Float16* aHt  = (_Float16*)bump((size_t)B * tNpix * 2);
  _Float16* aWs  = (_Float16*)bump((size_t)B * sNpix * 2);
  _Float16* aHs  = (_Float16*)bump((size_t)B * sNpix * 2);
  float* Gct = (float*)bump((size_t)B * C * C * 4);
  float* Gcs = (float*)bump((size_t)B * C * C * 4);
  float* Gwt = (float*)bump((size_t)B * tNpix * 4);
  float* Ght = (float*)bump((size_t)B * tNpix * 4);
  float* Gws = (float*)bump((size_t)B * sNpix * 4);
  float* Ghs = (float*)bump((size_t)B * sNpix * 4);

  // 1) f16 copies + h<->w transposed copies + per-batch channel transposes
  cvt_f32_f16<<<4096, 256, 0, stream>>>(t, t16, tN);
  cvt_f32_f16<<<8192, 256, 0, stream>>>(s, s16, sN);
  transpose2d<<<4096, 256, 0, stream>>>(t16, tT16, B * C, TW, TW);
  transpose2d<<<8192, 256, 0, stream>>>(s16, sT16, B * C, SW, SW);
  transpose2d<<<4096, 256, 0, stream>>>(t16, tCt, B, C, tNpix);
  transpose2d<<<8192, 256, 0, stream>>>(s16, sCt, B, C, sNpix);

  // 2) gram matrices (W-gram reads the transposed copy; H-gram reads the original)
  gram_spatial<<<dim3(TW / 16, TW / 16, B), 256, 0, stream>>>(tT16, Gwt, C, TW);
  gram_spatial<<<dim3(TW / 16, TW / 16, B), 256, 0, stream>>>(t16,  Ght, C, TW);
  gram_spatial<<<dim3(SW / 16, SW / 16, B), 256, 0, stream>>>(sT16, Gws, C, SW);
  gram_spatial<<<dim3(SW / 16, SW / 16, B), 256, 0, stream>>>(s16,  Ghs, C, SW);
  gram_xxt<<<dim3(C / 16, C / 16, B), 32, 0, stream>>>(t16, Gct, C, tNpix);
  gram_xxt<<<dim3(C / 16, C / 16, B), 32, 0, stream>>>(s16, Gcs, C, sNpix);

  // 3) softmaxes (spatial -> transposed f16 attn; channel -> row-major f16 attn)
  softmax_rows<<<B * TW, 256, 0, stream>>>(Gwt, aWt, TW,  1.0f, 1);
  softmax_rows<<<B * TW, 256, 0, stream>>>(Ght, aHt, TW,  1.0f, 1);
  softmax_rows<<<B * SW, 256, 0, stream>>>(Gws, aWs, SW,  1.0f, 1);
  softmax_rows<<<B * SW, 256, 0, stream>>>(Ghs, aHs, SW,  1.0f, 1);
  softmax_rows<<<B * C,  256, 0, stream>>>(Gct, aCt, C,  -1.0f, 0);
  softmax_rows<<<B * C,  256, 0, stream>>>(Gcs, aCs, C,  -1.0f, 0);

  // 4) applies, fused into d_out:  fused = 4*x + a*Yw + b*Yh + c*Yc + g*Yx
  apply_spatial<<<dim3(C * TW / 16, TW / 16, B), 32, 0, stream>>>(
      t16, aWt, t, out_t, sc_a, C * TW, TW, 0);
  apply_spatial<<<dim3(C * TW / 16, TW / 16, B), 32, 0, stream>>>(
      tT16, aHt, t, out_t, sc_b, C * TW, TW, 1);
  apply_channel<<<dim3(C / 16, tNpix / 16, B), 32, 0, stream>>>(
      aCt, aCs, tCt, out_t, sc_c, sc_g, C, tNpix);

  apply_spatial<<<dim3(C * SW / 16, SW / 16, B), 32, 0, stream>>>(
      s16, aWs, s, out_s, sc_d, C * SW, SW, 0);
  apply_spatial<<<dim3(C * SW / 16, SW / 16, B), 32, 0, stream>>>(
      sT16, aHs, s, out_s, sc_e, C * SW, SW, 1);
  apply_channel<<<dim3(C / 16, sNpix / 16, B), 32, 0, stream>>>(
      aCs, aCt, sCt, out_s, sc_f, sc_h, C, sNpix);
}
